// GRU_28638841929911
// MI455X (gfx1250) — matrix operand order repaired and checked by reading, not compiled, not run on hardware
//
#include <hip/hip_runtime.h>
#include <stdint.h>

typedef __attribute__((ext_vector_type(16))) __bf16 v16bf;
typedef __attribute__((ext_vector_type(8)))  float  v8f;

#define WMMA_BF16(a,b,c) __builtin_amdgcn_wmma_f32_16x16x32_bf16(false,(a),false,(b),(short)0,(c),false,false)

__device__ __forceinline__ unsigned short f2bf(float f){
    unsigned u = __float_as_uint(f);
    unsigned r = u + 0x7FFFu + ((u >> 16) & 1u);   // round-to-nearest-even
    return (unsigned short)(r >> 16);
}
__device__ __forceinline__ float bf2f(unsigned short b){
    return __uint_as_float(((unsigned)b) << 16);
}
__device__ __forceinline__ float sigmoid_f(float x){ return 1.f / (1.f + __expf(-x)); }
__device__ __forceinline__ float tanh_f(float x){ float t = __expf(-2.f*x); return (1.f - t)/(1.f + t); }

// A-matrix fragment (16x32 bf16), lanes0-15 hold K k0..k0+15 contiguous
__device__ __forceinline__ v16bf frag_a_lds(const unsigned short* base, int S, int row0, int k0, int lane){
    int half = lane >> 4, idx = lane & 15;
    const unsigned short* p = base + (row0 + idx) * S + k0 + half * 16;
    union { uint4 q[2]; v16bf v; } u;
    u.q[0] = *(const uint4*)(p);
    u.q[1] = *(const uint4*)(p + 8);
    return u.v;
}
// B-matrix fragment (32x16 bf16), column = lane&15, lanes0-15 hold K k0..k0+15 contiguous
__device__ __forceinline__ v16bf frag_b_lds(const unsigned short* base, int S, int n0, int k0, int lane){
    int half = lane >> 4, idx = lane & 15;
    const unsigned short* p = base + (n0 + idx) * S + k0 + half * 16;
    union { uint4 q[2]; v16bf v; } u;
    u.q[0] = *(const uint4*)(p);
    u.q[1] = *(const uint4*)(p + 8);
    return u.v;
}
__device__ __forceinline__ v16bf frag_b_glb(const unsigned short* __restrict__ base, int S, int n0, int k0, int lane){
    int half = lane >> 4, idx = lane & 15;
    const unsigned short* p = base + (n0 + idx) * S + k0 + half * 16;
    union { uint4 q[2]; v16bf v; } u;
    u.q[0] = *(const uint4*)(p);
    u.q[1] = *(const uint4*)(p + 8);
    return u.v;
}

__global__ void k_zero(float* __restrict__ p, long n){
    long i = (long)blockIdx.x * blockDim.x + threadIdx.x;
    long s = (long)gridDim.x * blockDim.x;
    for (; i < n; i += s) p[i] = 0.f;
}

__global__ void k_cvt(const float* __restrict__ Wz, const float* __restrict__ Wr,
                      const float* __restrict__ Ur, const float* __restrict__ Wh,
                      unsigned short* __restrict__ oWz, unsigned short* __restrict__ oWr,
                      unsigned short* __restrict__ oUr, unsigned short* __restrict__ oWh){
    int i = blockIdx.x * blockDim.x + threadIdx.x;
    int s = gridDim.x * blockDim.x;
    for (int j = i; j < 128 * 256; j += s){ oWz[j] = f2bf(Wz[j]); oWh[j] = f2bf(Wh[j]); }
    for (int j = i; j < 128 * 128; j += s){ oWr[j] = f2bf(Wr[j]); oUr[j] = f2bf(Ur[j]); }
}

// pre_r[e,o] = sum_i fmess[e,i]*W_r[o,i] + U_r_b[o]   (step-invariant)
__global__ void k_prer(const float* __restrict__ fmess, const unsigned short* __restrict__ Wr_bf,
                       const float* __restrict__ Urb, float* __restrict__ pre_r, int E){
    extern __shared__ char smem[];
    unsigned short* sWr = (unsigned short*)smem;               // [128][136]
    unsigned short* sA  = sWr + 128 * 136;                     // [16][136]
    const int t = threadIdx.x, w = t >> 5, lane = t & 31;

    for (int c = t; c < 128 * 16; c += 128){
        int r = c >> 4, j = c & 15;
        ((uint4*)(sWr + r * 136))[j] = ((const uint4*)Wr_bf)[r * 16 + j];
    }
    __syncthreads();

    int ntiles = (E + 15) / 16;
    for (int tile = blockIdx.x; tile < ntiles; tile += gridDim.x){
        int e0 = tile * 16;
        __syncthreads();
        for (int idx = t; idx < 16 * 128; idx += 128){
            int r = idx >> 7, c = idx & 127;
            int e = e0 + r; if (e >= E) e = E - 1;
            sA[r * 136 + c] = f2bf(fmess[(long)e * 128 + c]);
        }
        __syncthreads();

        int n0a = w * 16, n0b = (w + 4) * 16;
        v8f C0 = {}, C1 = {};
#pragma unroll
        for (int k = 0; k < 4; k++){
            v16bf a  = frag_a_lds(sA,  136, 0,   k * 32, lane);
            v16bf b0 = frag_b_lds(sWr, 136, n0a, k * 32, lane);
            v16bf b1 = frag_b_lds(sWr, 136, n0b, k * 32, lane);
            C0 = WMMA_BF16(a, b0, C0);
            C1 = WMMA_BF16(a, b1, C1);
        }
        int li = lane & 15, mb = (lane >> 4) * 8;
        float bb0 = Urb[n0a + li], bb1 = Urb[n0b + li];
        union { v8f v; float f[8]; } u0, u1; u0.v = C0; u1.v = C1;
#pragma unroll
        for (int j = 0; j < 8; j++){
            int e = e0 + mb + j;
            if (e < E){
                pre_r[(long)e * 128 + n0a + li] = u0.f[j] + bb0;
                pre_r[(long)e * 128 + n0b + li] = u1.f[j] + bb1;
            }
        }
    }
}

__global__ void k_step(const float* __restrict__ h_prev, const float* __restrict__ fmess,
                       const int* __restrict__ bgraph,
                       const unsigned short* __restrict__ Wz_bf,
                       const unsigned short* __restrict__ Ur_bf,
                       const unsigned short* __restrict__ Wh_bf,
                       const float* __restrict__ Wzb, const float* __restrict__ Whb,
                       const float* __restrict__ pre_r,
                       float* __restrict__ h_out, int E){
    extern __shared__ char smem[];
    unsigned short* sUr  = (unsigned short*)smem;      // [128][136]
    unsigned short* sHn  = sUr + 128 * 136;            // [128][136]  rows = e_loc*8 + k
    unsigned short* sZin = sHn + 128 * 136;            // [16][264]   cols 0-127 fmess, 128-255 sum(_gated)_h
    float* sZ    = (float*)(sZin + 16 * 264);          // [16][128]
    float* sSumH = sZ + 16 * 128;                      // [16][128]
    const int t = threadIdx.x, w = t >> 5, lane = t & 31;
    const int li = lane & 15, half = lane >> 4;

    for (int c = t; c < 128 * 16; c += 128){
        int r = c >> 4, j = c & 15;
        ((uint4*)(sUr + r * 136))[j] = ((const uint4*)Ur_bf)[r * 16 + j];
    }
    __syncthreads();

    int ntiles = (E + 15) / 16;
    for (int tile = blockIdx.x; tile < ntiles; tile += gridDim.x){
        int e0 = tile * 16;
        __syncthreads();   // protect LDS reuse vs previous iteration

        // ---- stage 1: stage fmess tile, gather neighbor h rows, build sum_h ----
        for (int idx = t; idx < 16 * 128; idx += 128){
            int r = idx >> 7, c = idx & 127;
            int e = e0 + r; if (e >= E) e = E - 1;
            sZin[r * 264 + c] = f2bf(fmess[(long)e * 128 + c]);
        }
        for (int i = 0; i < 128; i++){
            int e = e0 + (i >> 3); if (e >= E) e = E - 1;
            int src = bgraph[e * 8 + (i & 7)];
            sHn[i * 136 + t] = f2bf(h_prev[(long)src * 128 + t]);
        }
        __syncthreads();
        for (int idx = t; idx < 16 * 128; idx += 128){
            int r = idx >> 7, c = idx & 127;
            float s = 0.f;
#pragma unroll
            for (int kk = 0; kk < 8; kk++) s += bf2f(sHn[(r * 8 + kk) * 136 + c]);
            sSumH[r * 128 + c] = s;
            sZin[r * 264 + 128 + c] = f2bf(s);
        }
        __syncthreads();

        // ---- stage 2: z = sigmoid([fmess|sum_h] @ Wz^T + b) ----
        {
            int n0a = w * 16, n0b = (w + 4) * 16;
            v8f C0 = {}, C1 = {};
#pragma unroll
            for (int k = 0; k < 8; k++){
                v16bf a  = frag_a_lds(sZin, 264, 0,   k * 32, lane);
                v16bf b0 = frag_b_glb(Wz_bf, 256, n0a, k * 32, lane);
                v16bf b1 = frag_b_glb(Wz_bf, 256, n0b, k * 32, lane);
                C0 = WMMA_BF16(a, b0, C0);
                C1 = WMMA_BF16(a, b1, C1);
            }
            int mb = half * 8;
            float b0v = Wzb[n0a + li], b1v = Wzb[n0b + li];
            union { v8f v; float f[8]; } u0, u1; u0.v = C0; u1.v = C1;
#pragma unroll
            for (int j = 0; j < 8; j++){
                sZ[(mb + j) * 128 + n0a + li] = sigmoid_f(u0.f[j] + b0v);
                sZ[(mb + j) * 128 + n0b + li] = sigmoid_f(u1.f[j] + b1v);
            }
        }
        __syncthreads();

        // ---- stage 3: r2 = h_nei @ Ur^T ; gated reduction over 8 neighbors (in-register) ----
        {
            int m0a = w * 32, m0b = w * 32 + 16;
            for (int n = 0; n < 8; n++){
                int n0 = n * 16;
                v8f C0 = {}, C1 = {};
#pragma unroll
                for (int k = 0; k < 4; k++){
                    v16bf b  = frag_b_lds(sUr, 136, n0,  k * 32, lane);
                    v16bf a0 = frag_a_lds(sHn, 136, m0a, k * 32, lane);
                    v16bf a1 = frag_a_lds(sHn, 136, m0b, k * 32, lane);
                    C0 = WMMA_BF16(a0, b, C0);
                    C1 = WMMA_BF16(a1, b, C1);
                }
                int col = n0 + li;
                union { v8f v; float f[8]; } u0, u1; u0.v = C0; u1.v = C1;
                {   // C0: lanes0-15 -> edge 4w (k=0..7), lanes16-31 -> edge 4w+1
                    int el = 4 * w + half;
                    int e = e0 + el; if (e >= E) e = E - 1;
                    float pr = pre_r[(long)e * 128 + col];
                    float s = 0.f;
#pragma unroll
                    for (int j = 0; j < 8; j++){
                        float hn = bf2f(sHn[(el * 8 + j) * 136 + col]);
                        s += sigmoid_f(pr + u0.f[j]) * hn;
                    }
                    sZin[el * 264 + 128 + col] = f2bf(s);
                }
                {   // C1: edges 4w+2 / 4w+3
                    int el = 4 * w + 2 + half;
                    int e = e0 + el; if (e >= E) e = E - 1;
                    float pr = pre_r[(long)e * 128 + col];
                    float s = 0.f;
#pragma unroll
                    for (int j = 0; j < 8; j++){
                        float hn = bf2f(sHn[(el * 8 + j) * 136 + col]);
                        s += sigmoid_f(pr + u1.f[j]) * hn;
                    }
                    sZin[el * 264 + 128 + col] = f2bf(s);
                }
            }
        }
        __syncthreads();

        // ---- stage 4: pre_h = tanh([fmess|sum_gated_h] @ Wh^T + b); combine; store ----
        {
            int n0a = w * 16, n0b = (w + 4) * 16;
            v8f C0 = {}, C1 = {};
#pragma unroll
            for (int k = 0; k < 8; k++){
                v16bf a  = frag_a_lds(sZin, 264, 0,   k * 32, lane);
                v16bf b0 = frag_b_glb(Wh_bf, 256, n0a, k * 32, lane);
                v16bf b1 = frag_b_glb(Wh_bf, 256, n0b, k * 32, lane);
                C0 = WMMA_BF16(a, b0, C0);
                C1 = WMMA_BF16(a, b1, C1);
            }
            int mb = half * 8;
            float b0v = Whb[n0a + li], b1v = Whb[n0b + li];
            union { v8f v; float f[8]; } u0, u1; u0.v = C0; u1.v = C1;
#pragma unroll
            for (int j = 0; j < 8; j++){
                int el = mb + j, e = e0 + el;
                if (e < E){
                    float ph0 = tanh_f(u0.f[j] + b0v);
                    float ph1 = tanh_f(u1.f[j] + b1v);
                    float z0 = sZ[el * 128 + n0a + li], z1 = sZ[el * 128 + n0b + li];
                    float s0 = sSumH[el * 128 + n0a + li], s1 = sSumH[el * 128 + n0b + li];
                    float o0 = (e == 0) ? 0.f : ((1.f - z0) * s0 + z0 * ph0);
                    float o1 = (e == 0) ? 0.f : ((1.f - z1) * s1 + z1 * ph1);
                    h_out[(long)e * 128 + n0a + li] = o0;
                    h_out[(long)e * 128 + n0b + li] = o1;
                }
            }
        }
    }
}

extern "C" void kernel_launch(void* const* d_in, const int* in_sizes, int n_in,
                              void* d_out, int out_size, void* d_ws, size_t ws_size,
                              hipStream_t stream){
    const float* fmess = (const float*)d_in[0];
    const int*   bgr   = (const int*)d_in[1];
    const float* Wz    = (const float*)d_in[2];
    const float* Wzb   = (const float*)d_in[3];
    const float* Wr    = (const float*)d_in[4];
    const float* Ur    = (const float*)d_in[5];
    const float* Urb   = (const float*)d_in[6];
    const float* Wh    = (const float*)d_in[7];
    const float* Whb   = (const float*)d_in[8];
    const int E = in_sizes[0] / 128;
    float* h_out = (float*)d_out;

    char* ws = (char*)d_ws;
    float* hA   = (float*)ws;                  ws += (size_t)E * 128 * sizeof(float);
    float* preR = (float*)ws;                  ws += (size_t)E * 128 * sizeof(float);
    unsigned short* Wz_bf = (unsigned short*)ws; ws += 128 * 256 * 2;
    unsigned short* Wh_bf = (unsigned short*)ws; ws += 128 * 256 * 2;
    unsigned short* Ur_bf = (unsigned short*)ws; ws += 128 * 128 * 2;
    unsigned short* Wr_bf = (unsigned short*)ws; ws += 128 * 128 * 2;

    k_cvt<<<96, 256, 0, stream>>>(Wz, Wr, Ur, Wh, Wz_bf, Wr_bf, Ur_bf, Wh_bf);
    k_zero<<<2048, 256, 0, stream>>>(hA, (long)E * 128);

    int ntiles = (E + 15) / 16;
    int grid = ntiles < 1024 ? ntiles : 1024;
    size_t lds_pr = (size_t)(128 * 136 + 16 * 136) * 2;                              // 39,168 B
    k_prer<<<grid, 128, lds_pr, stream>>>(fmess, Wr_bf, Urb, preR, E);

    size_t lds_st = (size_t)(128 * 136 * 2 + 16 * 264) * 2 + (size_t)(16 * 128) * 4 * 2; // 94,464 B
    const float* hin = hA;
    for (int d = 0; d < 5; ++d){
        float* ho = (d % 2 == 0) ? h_out : hA;   // step 5 (d=4) lands in d_out
        k_step<<<grid, 128, lds_st, stream>>>(hin, fmess, bgr, Wz_bf, Ur_bf, Wh_bf,
                                              Wzb, Whb, preR, ho, E);
        hin = ho;
    }
}
